// IMP_GCN_71416716198487
// MI455X (gfx1250) — matrix operand order repaired and checked by reading, not compiled, not run on hardware
//
#include <hip/hip_runtime.h>

#define EMB 64
#define GROUPS 3
#define NLAYERS 3

typedef __attribute__((ext_vector_type(16))) _Float16 v16h;
typedef __attribute__((ext_vector_type(8)))  _Float16 h8v;    // 16 B, one b128
typedef __attribute__((ext_vector_type(8)))  float    v8f;

// ---------------------------------------------------------------------------
// Deterministic hash-based dropout (stand-in for jax.random.bernoulli key=42).
// ---------------------------------------------------------------------------
__device__ __forceinline__ unsigned drop_hash(unsigned x) {
  x ^= 42u * 0x9E3779B9u;
  x *= 0x85EBCA6Bu; x ^= x >> 13;
  x *= 0xC2B2AE35u; x ^= x >> 16;
  return x;
}

// ---------------------------------------------------------------------------
// ego = concat(user_emb, item_emb)
// ---------------------------------------------------------------------------
__global__ void k_concat(const float* __restrict__ ue, const float* __restrict__ ie,
                         float* __restrict__ ego, long split, long total) {
  long i = (long)blockIdx.x * 256 + threadIdx.x;
  if (i >= total) return;
  ego[i] = (i < split) ? ue[i] : ie[i - split];
}

// ---------------------------------------------------------------------------
// y[row[e]] += val[e] * x[col[e]]   (16 lanes per edge, float4 per lane)
// ---------------------------------------------------------------------------
__global__ void k_spmm(const float* __restrict__ val, const int* __restrict__ row,
                       const int* __restrict__ col, const float* __restrict__ x,
                       float* __restrict__ y, int nedges) {
  int t = blockIdx.x * 256 + threadIdx.x;
  int e = t >> 4;
  if (e >= nedges) return;
  float v = val[e];
  if (v == 0.0f) return;
  int c = col[e], r = row[e];
  int j = (t & 15) * 4;
  const float4 xv = *(const float4*)(x + (size_t)c * EMB + j);
  float* yd = y + (size_t)r * EMB + j;
  atomicAdd(yd + 0, v * xv.x);
  atomicAdd(yd + 1, v * xv.y);
  atomicAdd(yd + 2, v * xv.z);
  atomicAdd(yd + 3, v * xv.w);
}

// Gated variant: weight = val * gm[row][g] * gm[col][g]  (diag(g) A diag(g))
__global__ void k_spmm_gated(const float* __restrict__ val, const int* __restrict__ row,
                             const int* __restrict__ col, const float* __restrict__ gm,
                             int g, const float* __restrict__ x, float* __restrict__ y,
                             int nedges) {
  int t = blockIdx.x * 256 + threadIdx.x;
  int e = t >> 4;
  if (e >= nedges) return;
  int c = col[e], r = row[e];
  float v = val[e] * gm[(size_t)r * GROUPS + g] * gm[(size_t)c * GROUPS + g];
  if (v == 0.0f) return;                  // uniform per 16-lane edge group
  int j = (t & 15) * 4;
  const float4 xv = *(const float4*)(x + (size_t)c * EMB + j);
  float* yd = y + (size_t)r * EMB + j;
  atomicAdd(yd + 0, v * xv.x);
  atomicAdd(yd + 1, v * xv.y);
  atomicAdd(yd + 2, v * xv.z);
  atomicAdd(yd + 3, v * xv.w);
}

// ---------------------------------------------------------------------------
// Group assignment:
//   h  = dropout(leaky_relu((ego+side) @ W1 + b1))   [WMMA f32_16x16x32_f16]
//   s  = h @ Wg + bg ; gm = (s == rowmax(s)), items forced to all-ones.
// 256 threads = 8 waves; each wave owns a 16-row x 64-col tile.
// All LDS fragment traffic uses 16B-aligned b128 accesses:
//   sX  [per-wave 16x64 f16, row stride 72]  A-frag = 2x ds_load_b128
//   sWt [W1 transposed f16, row stride 72]   B-frag = 2x ds_load_b128
// ---------------------------------------------------------------------------
__global__ __launch_bounds__(256) void k_group_assign(
    const float* __restrict__ ego, const float* __restrict__ side,
    const float* __restrict__ W1, const float* __restrict__ bias1,
    const float* __restrict__ Wg, const float* __restrict__ bg,
    float* __restrict__ gm, int n_users, int N) {
  const int SXS = 72;   // f16 row stride: 144 B (16B aligned, conflict-free)
  const int SWS = 72;   // f16 row stride for transposed W
  const int SHS = 68;   // f32 row stride: 272 B (16B aligned, conflict-free)

  __shared__ _Float16 sWt[EMB * SWS];        // W1^T as f16: sWt[n][k]
  __shared__ _Float16 sX[8][16 * SXS];
  __shared__ float    sH[8][16 * SHS];

  const int tid  = threadIdx.x;
  const int w    = tid >> 5;
  const int lane = tid & 31;
  const int m16  = lane & 15;
  const int hi   = lane >> 4;
  const int rowBase = blockIdx.x * 128 + w * 16;

  // ---- Stage W1^T -> LDS f16. Thread: fixed n = tid&63, k-chunk of 16.
  // Global reads coalesced along n; two b128 LDS stores per thread.
  {
    int n  = tid & 63;
    int kc = (tid >> 6) * 16;
    h8v lo, hv;
    #pragma unroll
    for (int j = 0; j < 8; ++j) lo[j] = (_Float16)W1[(kc + j) * EMB + n];
    #pragma unroll
    for (int j = 0; j < 8; ++j) hv[j] = (_Float16)W1[(kc + 8 + j) * EMB + n];
    *(h8v*)&sWt[n * SWS + kc]     = lo;
    *(h8v*)&sWt[n * SWS + kc + 8] = hv;
  }

  // ---- Stage this wave's X tile = ego + side (zero-padded past N).
  // 8-col chunks: 2x float4 global loads -> one b128 LDS store.
  for (int ch = lane; ch < 16 * 8; ch += 32) {
    int r  = ch >> 3;
    int c8 = (ch & 7) * 8;
    h8v xv = {};
    int node = rowBase + r;
    if (node < N) {
      const float4* pe = (const float4*)(ego  + (size_t)node * EMB + c8);
      const float4* ps = (const float4*)(side + (size_t)node * EMB + c8);
      float4 e0 = pe[0], e1 = pe[1], s0 = ps[0], s1 = ps[1];
      xv[0] = (_Float16)(e0.x + s0.x); xv[1] = (_Float16)(e0.y + s0.y);
      xv[2] = (_Float16)(e0.z + s0.z); xv[3] = (_Float16)(e0.w + s0.w);
      xv[4] = (_Float16)(e1.x + s1.x); xv[5] = (_Float16)(e1.y + s1.y);
      xv[6] = (_Float16)(e1.z + s1.z); xv[7] = (_Float16)(e1.w + s1.w);
    }
    *(h8v*)&sX[w][r * SXS + c8] = xv;
  }
  __syncthreads();

  // ---- 16x64 tile: 4 accumulators, K = 64 in two K=32 WMMA steps.
  v8f acc[4] = {};
  const _Float16* xrow = &sX[w][m16 * SXS];
  #pragma unroll
  for (int kb = 0; kb < EMB; kb += 32) {
    // A fragment: per-lane K = {kb+16hi..+15} (2x b128)
    h8v alo = *(const h8v*)(xrow + kb + 16 * hi);
    h8v ahi = *(const h8v*)(xrow + kb + 16 * hi + 8);
    v16h a = __builtin_shufflevector(alo, ahi, 0, 1, 2, 3, 4, 5, 6, 7,
                                     8, 9, 10, 11, 12, 13, 14, 15);
    #pragma unroll
    for (int nb = 0; nb < 4; ++nb) {
      // B fragment: per-lane K = {kb+16hi..+15} at col n = m16+16nb (2x b128)
      const _Float16* wrow = &sWt[(m16 + nb * 16) * SWS + kb + 16 * hi];
      h8v blo = *(const h8v*)(wrow);
      h8v bhi = *(const h8v*)(wrow + 8);
      v16h b = __builtin_shufflevector(blo, bhi, 0, 1, 2, 3, 4, 5, 6, 7,
                                       8, 9, 10, 11, 12, 13, 14, 15);
      acc[nb] = __builtin_amdgcn_wmma_f32_16x16x32_f16(
          false, a, false, b, (short)0, acc[nb], false, false);
    }
  }

  // ---- Epilogue: bias + leaky_relu + dropout(p=0.6) -> sH
  #pragma unroll
  for (int nb = 0; nb < 4; ++nb) {
    int n = m16 + nb * 16;
    float bv = bias1[n];
    #pragma unroll
    for (int i = 0; i < 8; ++i) {
      int r = i + 8 * hi;                       // C/D layout: M = i + 8*hi
      float v = acc[nb][i] + bv;
      v = v > 0.0f ? v : 0.01f * v;
      unsigned hsh = drop_hash((unsigned)((rowBase + r) * EMB + n));
      v = ((hsh >> 8) * (1.0f / 16777216.0f) < 0.4f) ? v * 2.5f : 0.0f;
      sH[w][r * SHS + n] = v;
    }
  }
  __syncthreads();

  // ---- Scorer + argmax mask: one lane per row, float4 LDS reads.
  if (lane < 16) {
    int node = rowBase + m16;
    if (node < N) {
      if (node >= n_users) {
        gm[(size_t)node * GROUPS + 0] = 1.0f;
        gm[(size_t)node * GROUPS + 1] = 1.0f;
        gm[(size_t)node * GROUPS + 2] = 1.0f;
      } else {
        float s0 = bg[0], s1 = bg[1], s2 = bg[2];
        const float4* hr4 = (const float4*)&sH[w][m16 * SHS];
        #pragma unroll
        for (int k4 = 0; k4 < 16; ++k4) {
          float4 hv = hr4[k4];
          int k = k4 * 4;
          s0 += hv.x * Wg[(k + 0) * GROUPS + 0] + hv.y * Wg[(k + 1) * GROUPS + 0] +
                hv.z * Wg[(k + 2) * GROUPS + 0] + hv.w * Wg[(k + 3) * GROUPS + 0];
          s1 += hv.x * Wg[(k + 0) * GROUPS + 1] + hv.y * Wg[(k + 1) * GROUPS + 1] +
                hv.z * Wg[(k + 2) * GROUPS + 1] + hv.w * Wg[(k + 3) * GROUPS + 1];
          s2 += hv.x * Wg[(k + 0) * GROUPS + 2] + hv.y * Wg[(k + 1) * GROUPS + 2] +
                hv.z * Wg[(k + 2) * GROUPS + 2] + hv.w * Wg[(k + 3) * GROUPS + 2];
        }
        float mx = fmaxf(s0, fmaxf(s1, s2));
        gm[(size_t)node * GROUPS + 0] = (s0 == mx) ? 1.0f : 0.0f;
        gm[(size_t)node * GROUPS + 1] = (s1 == mx) ? 1.0f : 0.0f;
        gm[(size_t)node * GROUPS + 2] = (s2 == mx) ? 1.0f : 0.0f;
      }
    }
  }
}

// ---------------------------------------------------------------------------
// out = ego + side ;  ego_g[0..2] = ego
// ---------------------------------------------------------------------------
__global__ void k_postinit(const float* __restrict__ ego, const float* __restrict__ side,
                           float* __restrict__ out, float* __restrict__ e0,
                           float* __restrict__ e1, float* __restrict__ e2, long total) {
  long i = (long)blockIdx.x * 256 + threadIdx.x;
  if (i >= total) return;
  float e = ego[i];
  out[i] = e + side[i];
  e0[i] = e; e1[i] = e; e2[i] = e;
}

__global__ void k_axpy(float* __restrict__ y, const float* __restrict__ x, long total) {
  long i = (long)blockIdx.x * 256 + threadIdx.x;
  if (i >= total) return;
  y[i] += x[i];
}

// ---------------------------------------------------------------------------
extern "C" void kernel_launch(void* const* d_in, const int* in_sizes, int n_in,
                              void* d_out, int out_size, void* d_ws, size_t ws_size,
                              hipStream_t stream) {
  const float* user_emb = (const float*)d_in[0];
  const float* item_emb = (const float*)d_in[1];
  const float* W1       = (const float*)d_in[2];
  const float* b1       = (const float*)d_in[3];
  const float* Wg       = (const float*)d_in[4];
  const float* bg       = (const float*)d_in[5];
  const float* eval     = (const float*)d_in[6];
  const int*   erow     = (const int*)d_in[7];
  const int*   ecol     = (const int*)d_in[8];

  const int n_users = in_sizes[0] / EMB;
  const int n_items = in_sizes[1] / EMB;
  const int NE      = in_sizes[6];
  const int N       = n_users + n_items;
  const long nd     = (long)N * EMB;

  // Workspace carving
  float* ego   = (float*)d_ws;
  float* side  = ego  + nd;
  float* eg0   = side + nd;
  float* eg1   = eg0  + nd;
  float* eg2   = eg1  + nd;
  float* sideg = eg2  + nd;
  float* gm    = sideg + nd;                    // N*GROUPS floats
  if (ws_size < (size_t)(6 * nd + (long)N * GROUPS) * sizeof(float)) return;

  float* out = (float*)d_out;                   // [N, EMB] = users then items
  float* egp[GROUPS] = { eg0, eg1, eg2 };

  const int ewGrid   = (int)((nd + 255) / 256);          // elementwise grids
  const int spmmGrid = (NE * 16 + 255) / 256;            // 16 lanes per edge
  const int gaGrid   = (N + 127) / 128;                  // 128 rows per block

  // 1) ego = concat(user, item)
  k_concat<<<ewGrid, 256, 0, stream>>>(user_emb, item_emb, ego,
                                       (long)n_users * EMB, nd);
  // 2) side = A_norm @ ego
  hipMemsetAsync(side, 0, nd * sizeof(float), stream);
  k_spmm<<<spmmGrid, 256, 0, stream>>>(eval, erow, ecol, ego, side, NE);
  // 3) group masks via WMMA GEMM + scorer
  k_group_assign<<<gaGrid, 256, 0, stream>>>(ego, side, W1, b1, Wg, bg,
                                             gm, n_users, N);
  // 4) all_emb = ego + side ; ego_g[g] = ego
  k_postinit<<<ewGrid, 256, 0, stream>>>(ego, side, out, eg0, eg1, eg2, nd);

  // 5) layers 1..NLAYERS-1: per group, gated propagate then re-diffuse
  for (int layer = 1; layer < NLAYERS; ++layer) {
    for (int g = 0; g < GROUPS; ++g) {
      hipMemsetAsync(sideg, 0, nd * sizeof(float), stream);
      k_spmm_gated<<<spmmGrid, 256, 0, stream>>>(eval, erow, ecol, gm, g,
                                                 egp[g], sideg, NE);
      k_spmm<<<spmmGrid, 256, 0, stream>>>(eval, erow, ecol, sideg, out, NE);
      k_axpy<<<ewGrid, 256, 0, stream>>>(egp[g], sideg, nd);
    }
  }
}
